// ScaledDotProductAttention_1460288880892
// MI455X (gfx1250) — compile-verified, matrix operand order checked by reading, not run on hardware
//
#include <hip/hip_runtime.h>

// ---------------------------------------------------------------------------
// Flash-attention forward for B=2,H=16,S=2048,D=128 (causal), fp32 I/O,
// bf16 WMMA compute (v_wmma_f32_16x16x32_bf16), fp32 accumulate + softmax.
// Block = 256 threads = 8 wave32; each block owns a 128-row Q tile.
// Double-buffered K/V staging in LDS; packed f32->bf16 conversion.
// ---------------------------------------------------------------------------

typedef __attribute__((ext_vector_type(16))) __bf16 v16bf;
typedef __attribute__((ext_vector_type(8)))  __bf16 v8bf;
typedef __attribute__((ext_vector_type(4)))  __bf16 v4bf;
typedef __attribute__((ext_vector_type(8)))  float  v8f;
typedef __attribute__((ext_vector_type(4)))  float  v4f;

#define B_   2
#define H_   16
#define S_   2048
#define D_   128
#define QT   128            // query rows per block
#define KT   32             // keys per inner iteration
#define KSTR (D_ + 8)       // LDS K tile row stride (bf16 elems) -> 272B, conflict-free
#define VSTR (KT + 8)       // LDS V^T row stride -> 80B
#define PSTR (KT + 8)       // per-wave P scratch row stride -> 80B

__global__ __launch_bounds__(256, 1)
void fa_fwd(const float* __restrict__ Qg, const float* __restrict__ Kg,
            const float* __restrict__ Vg, float* __restrict__ Og)
{
  __shared__ __align__(16) __bf16 sK [2][KT * KSTR];     // 2 x  8704 B : K tile  [key][d]
  __shared__ __align__(16) __bf16 sVt[2][D_ * VSTR];     // 2 x 10240 B : V tile^T [d][key]
  __shared__ __align__(16) __bf16 sP [8 * 16 * PSTR];    //     10240 B : per-wave P [row][key]

  const int tid  = threadIdx.x;
  const int wv   = tid >> 5;      // wave 0..7
  const int lane = tid & 31;
  const int l16  = lane & 15;
  const int hi   = lane >> 4;     // lane half (selects K sub-chunk in A/B frags)

  const int qb = blockIdx.x, h = blockIdx.y, b = blockIdx.z;
  const int qBase = qb * QT;
  const size_t hoff = ((size_t)(b * H_ + h)) * (size_t)S_ * D_;
  const float* Qh = Qg + hoff;
  const float* Kh = Kg + hoff;
  const float* Vh = Vg + hoff;
  float*       Oh = Og + hoff;

  // 1/sqrt(128) * log2(e): softmax done in exp2 domain
  const float sc = 0.08838834764831845f * 1.4426950408889634f;

  // cooperative K/V tile staging: K row-major bf16, V transposed bf16
  auto stageKV = [&](int kt0, int buf) {
    #pragma unroll
    for (int rep = 0; rep < (KT * D_) / (4 * 256); ++rep) {
      const int idx = rep * 256 + tid;
      const int k  = idx >> 5;          // / (D_/4)
      const int d4 = idx & 31;
      const v4f kv = *(const v4f*)(Kh + (size_t)(kt0 + k) * D_ + d4 * 4);
      const v4bf kb = __builtin_convertvector(kv, v4bf);   // v_cvt_pk_bf16_f32 x2
      *(v4bf*)(&sK[buf][k * KSTR + d4 * 4]) = kb;          // single ds_store_b64
      const v4f vvf = *(const v4f*)(Vh + (size_t)(kt0 + k) * D_ + d4 * 4);
      const v4bf vb = __builtin_convertvector(vvf, v4bf);
      const int d0 = d4 * 4;
      sVt[buf][(d0 + 0) * VSTR + k] = vb[0];
      sVt[buf][(d0 + 1) * VSTR + k] = vb[1];
      sVt[buf][(d0 + 2) * VSTR + k] = vb[2];
      sVt[buf][(d0 + 3) * VSTR + k] = vb[3];
    }
  };

  // ---- Q A-fragments (16x32 bf16 each, 4 chunks cover D=128), kept in VGPRs.
  // A layout: row M = lane%16; lanes<16 hold K = c*32 + {0..7, 16..23},
  //           lanes>=16 hold K = c*32 + {8..15, 24..31}.
  v16bf qa[4];
  {
    const int qRow = qBase + wv * 16 + l16;
    const float* qp = Qh + (size_t)qRow * D_;
    #pragma unroll
    for (int c = 0; c < 4; ++c) {
      const float* src = qp + c * 32 + hi * 8;
      const v4bf b0 = __builtin_convertvector(*(const v4f*)(src),      v4bf);
      const v4bf b1 = __builtin_convertvector(*(const v4f*)(src + 4),  v4bf);
      const v4bf b2 = __builtin_convertvector(*(const v4f*)(src + 16), v4bf);
      const v4bf b3 = __builtin_convertvector(*(const v4f*)(src + 20), v4bf);
      const v8bf lo = __builtin_shufflevector(b0, b1, 0,1,2,3,4,5,6,7);
      const v8bf hh = __builtin_shufflevector(b2, b3, 0,1,2,3,4,5,6,7);
      qa[c] = __builtin_shufflevector(lo, hh, 0,1,2,3,4,5,6,7,8,9,10,11,12,13,14,15);
    }
  }

  const v8f zero8 = {0.f, 0.f, 0.f, 0.f, 0.f, 0.f, 0.f, 0.f};
  v8f o[8];                       // O accumulator: 16 rows x 128 cols (8 C-tiles)
  #pragma unroll
  for (int t = 0; t < 8; ++t) o[t] = zero8;
  float m[8], lsum[8];            // per-row running max / running sum
  #pragma unroll
  for (int i = 0; i < 8; ++i) { m[i] = -3.0e38f; lsum[i] = 0.0f; }

  const int nkt      = (qBase + QT) / KT;   // causal: keys up to end of Q tile
  const int qWaveMin = qBase + wv * 16;     // smallest query row of this wave

  // ---- software pipeline: stage tile 0, then overlap stage(j+1) + compute(j)
  stageKV(0, 0);
  int p = 0;

  for (int j = 0; j < nkt; ++j) {
    const int kt0 = j * KT;
    __syncthreads();                       // buf p staged; prior compute done

    if (j + 1 < nkt) stageKV((j + 1) * KT, p ^ 1);   // overlaps with WMMAs below
    if (j + 2 < nkt) {                                // global_prefetch_b8 two ahead
      const size_t nb = (size_t)(kt0 + 2 * KT) * D_;
      __builtin_prefetch(Kh + nb + tid * 16, 0, 1);
      __builtin_prefetch(Vh + nb + tid * 16, 0, 1);
    }

    // waves whose rows are all above this key tile (fully masked) skip compute
    if (kt0 <= qWaveMin + 15) {
      // ---- scores = Q * K^T : two 16x16 C-tiles (32 keys), K=32 per WMMA ----
      v8f a0 = zero8, a1 = zero8;
      #pragma unroll
      for (int c = 0; c < 4; ++c) {
        const __bf16* bp = &sK[p][l16 * KSTR + c * 32 + hi * 8];
        v8bf lo = *(const v8bf*)bp;
        v8bf hh = *(const v8bf*)(bp + 16);
        v16bf bf = __builtin_shufflevector(lo, hh, 0,1,2,3,4,5,6,7,8,9,10,11,12,13,14,15);
        a0 = __builtin_amdgcn_wmma_f32_16x16x32_bf16(false, qa[c], false, bf,
                                                     (short)0, a0, false, false);
      }
      #pragma unroll
      for (int c = 0; c < 4; ++c) {
        const __bf16* bp = &sK[p][(16 + l16) * KSTR + c * 32 + hi * 8];
        v8bf lo = *(const v8bf*)bp;
        v8bf hh = *(const v8bf*)(bp + 16);
        v16bf bf = __builtin_shufflevector(lo, hh, 0,1,2,3,4,5,6,7,8,9,10,11,12,13,14,15);
        a1 = __builtin_amdgcn_wmma_f32_16x16x32_bf16(false, qa[c], false, bf,
                                                     (short)0, a1, false, false);
      }

      // ---- scale + causal mask (C layout: row M = i + 8*hi, col N = l16) ----
      const bool needMask = (kt0 + KT - 1) > qWaveMin;
      float s0[8], s1[8];
      #pragma unroll
      for (int i = 0; i < 8; ++i) {
        float v0 = a0[i] * sc;
        float v1 = a1[i] * sc;
        if (needMask) {
          const int qr = qWaveMin + i + hi * 8;
          if (kt0 + l16      > qr) v0 = -1.0e30f;
          if (kt0 + 16 + l16 > qr) v1 = -1.0e30f;
        }
        s0[i] = v0; s1[i] = v1;
      }

      // ---- online softmax update + P -> LDS (bf16) --------------------------
      #pragma unroll
      for (int i = 0; i < 8; ++i) {
        float rm = fmaxf(s0[i], s1[i]);
        rm = fmaxf(rm, __shfl_xor(rm, 1, 32));
        rm = fmaxf(rm, __shfl_xor(rm, 2, 32));
        rm = fmaxf(rm, __shfl_xor(rm, 4, 32));
        rm = fmaxf(rm, __shfl_xor(rm, 8, 32));
        const float nm = fmaxf(m[i], rm);
        const float al = exp2f(m[i] - nm);
        m[i] = nm;
        const float p0 = exp2f(s0[i] - nm);
        const float p1 = exp2f(s1[i] - nm);
        float rs = p0 + p1;
        rs += __shfl_xor(rs, 1, 32);
        rs += __shfl_xor(rs, 2, 32);
        rs += __shfl_xor(rs, 4, 32);
        rs += __shfl_xor(rs, 8, 32);
        lsum[i] = lsum[i] * al + rs;
        #pragma unroll
        for (int t = 0; t < 8; ++t) o[t][i] *= al;
        __bf16* pw = &sP[(wv * 16 + i + hi * 8) * PSTR];
        pw[l16]      = (__bf16)p0;
        pw[16 + l16] = (__bf16)p1;
      }
      // wave-local LDS write->read ordering (DS ops are in-order per wave)
      asm volatile("s_wait_dscnt 0" ::: "memory");

      // ---- O += P(16x32) * V(32x128): one K=32 WMMA per 16-col output tile --
      const __bf16* pp = &sP[(wv * 16 + l16) * PSTR + hi * 8];
      v8bf plo = *(const v8bf*)pp;
      v8bf phi = *(const v8bf*)(pp + 16);
      v16bf pa = __builtin_shufflevector(plo, phi, 0,1,2,3,4,5,6,7,8,9,10,11,12,13,14,15);
      #pragma unroll
      for (int t = 0; t < 8; ++t) {
        const __bf16* vp = &sVt[p][(t * 16 + l16) * VSTR + hi * 8];
        v8bf vlo = *(const v8bf*)vp;
        v8bf vhi = *(const v8bf*)(vp + 16);
        v16bf vb = __builtin_shufflevector(vlo, vhi, 0,1,2,3,4,5,6,7,8,9,10,11,12,13,14,15);
        o[t] = __builtin_amdgcn_wmma_f32_16x16x32_bf16(false, pa, false, vb,
                                                       (short)0, o[t], false, false);
      }
    }
    p ^= 1;
  }

  // ---- normalize by 1/l and write out (fp32) ------------------------------
  float inv[8];
  #pragma unroll
  for (int i = 0; i < 8; ++i) inv[i] = 1.0f / lsum[i];
  const int qr0 = qBase + wv * 16 + hi * 8;
  #pragma unroll
  for (int i = 0; i < 8; ++i) {
    float* orow = Oh + (size_t)(qr0 + i) * D_ + l16;
    #pragma unroll
    for (int t = 0; t < 8; ++t) orow[t * 16] = o[t][i] * inv[i];
  }
}

extern "C" void kernel_launch(void* const* d_in, const int* in_sizes, int n_in,
                              void* d_out, int out_size, void* d_ws, size_t ws_size,
                              hipStream_t stream) {
  (void)in_sizes; (void)n_in; (void)out_size; (void)d_ws; (void)ws_size;
  const float* Q = (const float*)d_in[0];
  const float* K = (const float*)d_in[1];
  const float* V = (const float*)d_in[2];
  // d_in[3] is the causal mask; causality is computed analytically in-kernel.
  float* O = (float*)d_out;
  dim3 grid(S_ / QT, H_, B_);
  dim3 block(256);
  hipLaunchKernelGGL(fa_fwd, grid, block, 0, stream, Q, K, V, O);
}
